// FusedMoE_18408229831237
// MI455X (gfx1250) — matrix operand order repaired and checked by reading, not compiled, not run on hardware
//
#include <hip/hip_runtime.h>

// Problem constants (match reference)
constexpr int TOK  = 16384;   // tokens
constexpr int HID  = 2048;    // hidden
constexpr int NE   = 16;      // experts
constexpr int IDIM = 1408;    // intermediate

constexpr int BM   = 32;         // token tile (2 WMMA M-tiles)
constexpr int XSTR = HID  + 8;   // LDS row stride (halfwords), +16B pad
constexpr int ASTR = IDIM + 8;   // LDS act row stride (halfwords)

typedef __bf16          v16bf __attribute__((ext_vector_type(16)));
typedef unsigned short  v16us __attribute__((ext_vector_type(16)));
typedef unsigned int    v8u   __attribute__((ext_vector_type(8)));
typedef float           v8f   __attribute__((ext_vector_type(8)));

// fp32 -> bf16 round-to-nearest-even (low-volume paths)
__device__ __forceinline__ unsigned short f2bf(float x) {
    unsigned int u = __float_as_uint(x);
    u += 0x7fffu + ((u >> 16) & 1u);
    return (unsigned short)(u >> 16);
}

// Two fp32 -> packed bf16x2 (round-half-up) via v_perm_b32
__device__ __forceinline__ unsigned int pack_bf2(float lo, float hi) {
    unsigned int a = __float_as_uint(lo) + 0x8000u;
    unsigned int b = __float_as_uint(hi) + 0x8000u;
    return __builtin_amdgcn_perm(b, a, 0x07060302u);
}

// B-fragment: 16 contiguous fp32 -> bf16 (fallback path)
__device__ __forceinline__ v16us ldg16_cvt(const float* __restrict__ p) {
    v8u u;
#pragma unroll
    for (int j = 0; j < 8; ++j) u[j] = pack_bf2(p[2 * j], p[2 * j + 1]);
    return __builtin_bit_cast(v16us, u);
}

// B-fragment: 16 contiguous bf16 (fast path: 2x global_load_b128)
__device__ __forceinline__ v16us ldg16_us(const unsigned short* __restrict__ p) {
    v16us r;
#pragma unroll
    for (int j = 0; j < 16; ++j) r[j] = p[j];
    return r;
}

// A-fragment from an LDS bf16 row (contiguous k order: k = 16*half + j)
__device__ __forceinline__ v16us lds_afrag(const unsigned short* __restrict__ row, int half) {
    v16us r;
#pragma unroll
    for (int j = 0; j < 16; ++j) r[j] = row[half * 16 + j];
    return r;
}

__device__ __forceinline__ v8f wmma_bf16(v16us a, v16us b, v8f c) {
    return __builtin_amdgcn_wmma_f32_16x16x32_bf16(
        false, __builtin_bit_cast(v16bf, a),
        false, __builtin_bit_cast(v16bf, b),
        (short)0, c, false, false);
}

// ---------------------------------------------------------------------------
// Kernel 0: bulk fp32 -> bf16 conversion (bandwidth-bound)
// ---------------------------------------------------------------------------
__global__ void cvt_bf16(const float* __restrict__ s, unsigned int* __restrict__ d,
                         long long n4) {
    long long i      = (long long)blockIdx.x * blockDim.x + threadIdx.x;
    long long stride = (long long)gridDim.x * blockDim.x;
    for (; i < n4; i += stride) {
        float4 f = ((const float4*)s)[i];
        uint2 o;
        o.x = pack_bf2(f.x, f.y);
        o.y = pack_bf2(f.z, f.w);
        ((uint2*)d)[i] = o;
    }
}

// ---------------------------------------------------------------------------
// Kernel 1: router softmax + top-2 + renorm + per-expert compaction
// ---------------------------------------------------------------------------
__global__ void moe_router(const float* __restrict__ logits,
                           int* __restrict__ cnt,
                           int* __restrict__ tids,
                           float* __restrict__ twts) {
    int t = blockIdx.x * blockDim.x + threadIdx.x;
    if (t >= TOK) return;

    float l[NE];
    float mx = -__FLT_MAX__;
#pragma unroll
    for (int e = 0; e < NE; ++e) { l[e] = logits[t * NE + e]; mx = fmaxf(mx, l[e]); }

    int i0 = 0; float v0 = l[0];
#pragma unroll
    for (int e = 1; e < NE; ++e) if (l[e] > v0) { v0 = l[e]; i0 = e; }
    int i1 = (i0 == 0) ? 1 : 0; float v1 = l[i1];
#pragma unroll
    for (int e = 0; e < NE; ++e) if (e != i0 && l[e] > v1) { v1 = l[e]; i1 = e; }

    float p0 = __expf(v0 - mx), p1 = __expf(v1 - mx);
    float inv = __builtin_amdgcn_rcpf(p0 + p1);
    float w0 = p0 * inv, w1 = p1 * inv;

    int s0 = atomicAdd(&cnt[i0], 1);
    tids[i0 * TOK + s0] = t;  twts[i0 * TOK + s0] = w0;
    int s1 = atomicAdd(&cnt[i1], 1);
    tids[i1 * TOK + s1] = t;  twts[i1 * TOK + s1] = w1;
}

// ---------------------------------------------------------------------------
// Kernel 2: fused per-expert  act = silu(X w1^T) * (X w3^T);  out += w * act w2^T
// 32-token tile; whole X tile in LDS; each B-fragment feeds 2 WMMAs.
// ---------------------------------------------------------------------------
template <bool PRE>
__global__ __launch_bounds__(256) void moe_expert(
    const float* __restrict__ hsf,  const unsigned short* __restrict__ hsb,
    const float* __restrict__ w13f, const unsigned short* __restrict__ w13b,
    const float* __restrict__ w2f,  const unsigned short* __restrict__ w2b,
    const int* __restrict__ cnt,    const int* __restrict__ tids,
    const float* __restrict__ twts, float* __restrict__ out) {

    const int e    = blockIdx.y;
    const int tile = blockIdx.x;
    const int n_e  = cnt[e];
    if (tile * BM >= n_e) return;

    __shared__ int            sTok[BM];
    __shared__ float          sW[BM];
    __shared__ unsigned short sX[BM * XSTR];      // full X tile (bf16), ~128.5 KB
    __shared__ unsigned short sAct[BM * ASTR];    // fused intermediate (bf16), ~88.5 KB

    const int tid = threadIdx.x;
    if (tid < BM) {
        int r = tile * BM + tid;
        if (r < n_e) { sTok[tid] = tids[e * TOK + r]; sW[tid] = twts[e * TOK + r]; }
        else         { sTok[tid] = 0;                 sW[tid] = 0.0f; }
    }
    __syncthreads();

    // ---- stage entire X tile: 32 rows x 2048 bf16, 512B/thread, b128 copies
    {
        const int m  = tid >> 3;                   // row (8 threads per row)
        const int c8 = tid & 7;
        const size_t rowg = (size_t)sTok[m] * HID;
        uint4* dst = (uint4*)(&sX[m * XSTR]);
        if constexpr (PRE) {
            const uint4* src = (const uint4*)(hsb + rowg);   // 8 bf16 per uint4
#pragma unroll
            for (int r = 0; r < 32; ++r)
                dst[r * 8 + c8] = src[r * 8 + c8];
        } else {
            const float4* src = (const float4*)(hsf + rowg);
#pragma unroll
            for (int r = 0; r < 32; ++r) {
                float4 f0 = src[(r * 8 + c8) * 2];
                float4 f1 = src[(r * 8 + c8) * 2 + 1];
                uint4 o;
                o.x = pack_bf2(f0.x, f0.y); o.y = pack_bf2(f0.z, f0.w);
                o.z = pack_bf2(f1.x, f1.y); o.w = pack_bf2(f1.z, f1.w);
                dst[r * 8 + c8] = o;
            }
        }
    }
    __syncthreads();

    const int wave = tid >> 5;
    const int lane = tid & 31;
    const int half = lane >> 4;
    const int l16  = lane & 15;
    const unsigned short* xrow0 = &sX[l16 * XSTR];          // M-tile 0 (rows 0..15)
    const unsigned short* xrow1 = &sX[(16 + l16) * XSTR];   // M-tile 1 (rows 16..31)

    // ---------------- GEMM1 + SiLU-and-mul -> sAct ----------------
    for (int nt = 0; nt < IDIM / 128; ++nt) {            // 11 tiles of 16 cols per wave
        const int col = wave * (IDIM / 8) + nt * 16;
        v8f g1lo = {}, g1hi = {}, g3lo = {}, g3hi = {};

        const size_t r1 = (size_t)e * (2 * IDIM) * HID + (size_t)(col + l16) * HID + half * 16;
        const size_t r3 = r1 + (size_t)IDIM * HID;

        for (int k0 = 0; k0 < HID; k0 += 32) {
            v16us alo = lds_afrag(xrow0 + k0, half);
            v16us ahi = lds_afrag(xrow1 + k0, half);
            v16us b1, b3;
            if constexpr (PRE) {
                __builtin_prefetch(w13b + r1 + k0 + 32, 0, 1);   // global_prefetch_b8
                __builtin_prefetch(w13b + r3 + k0 + 32, 0, 1);
                b1 = ldg16_us(w13b + r1 + k0);
                b3 = ldg16_us(w13b + r3 + k0);
            } else {
                b1 = ldg16_cvt(w13f + r1 + k0);
                b3 = ldg16_cvt(w13f + r3 + k0);
            }
            g1lo = wmma_bf16(alo, b1, g1lo);
            g1hi = wmma_bf16(ahi, b1, g1hi);
            g3lo = wmma_bf16(alo, b3, g3lo);
            g3hi = wmma_bf16(ahi, b3, g3hi);
        }
        // C/D layout: element r -> M = half*8 + r, N = l16
#pragma unroll
        for (int r = 0; r < 8; ++r) {
            int m0 = half * 8 + r;
            float a = g1lo[r], b = g3lo[r];
            float sig = __builtin_amdgcn_rcpf(1.0f + __expf(-a));
            sAct[m0 * ASTR + col + l16] = f2bf(a * sig * b);
            float c = g1hi[r], d = g3hi[r];
            float sg2 = __builtin_amdgcn_rcpf(1.0f + __expf(-c));
            sAct[(16 + m0) * ASTR + col + l16] = f2bf(c * sg2 * d);
        }
    }
    __syncthreads();

    // ---------------- GEMM2: y = act @ w2^T ; out += w * y ----------------
    const unsigned short* arow0 = &sAct[l16 * ASTR];
    const unsigned short* arow1 = &sAct[(16 + l16) * ASTR];
    for (int ht = 0; ht < HID / 128; ++ht) {             // 16 tiles of 16 cols per wave
        const int ch = wave * (HID / 8) + ht * 16;
        v8f ylo = {}, yhi = {};
        const size_t r2 = (size_t)e * HID * IDIM + (size_t)(ch + l16) * IDIM + half * 16;
        for (int k0 = 0; k0 < IDIM; k0 += 32) {
            v16us alo = lds_afrag(arow0 + k0, half);
            v16us ahi = lds_afrag(arow1 + k0, half);
            v16us b;
            if constexpr (PRE) {
                __builtin_prefetch(w2b + r2 + k0 + 32, 0, 1);
                b = ldg16_us(w2b + r2 + k0);
            } else {
                b = ldg16_cvt(w2f + r2 + k0);
            }
            ylo = wmma_bf16(alo, b, ylo);
            yhi = wmma_bf16(ahi, b, yhi);
        }
#pragma unroll
        for (int r = 0; r < 8; ++r) {
            int m0 = half * 8 + r;
            atomicAdd(&out[(size_t)sTok[m0] * HID + ch + l16],      sW[m0]      * ylo[r]);
            atomicAdd(&out[(size_t)sTok[16 + m0] * HID + ch + l16], sW[16 + m0] * yhi[r]);
        }
    }
}

// ---------------------------------------------------------------------------
extern "C" void kernel_launch(void* const* d_in, const int* in_sizes, int n_in,
                              void* d_out, int out_size, void* d_ws, size_t ws_size,
                              hipStream_t stream) {
    (void)in_sizes; (void)n_in; (void)out_size;
    const float* hs  = (const float*)d_in[0];
    const float* lg  = (const float*)d_in[1];
    const float* w13 = (const float*)d_in[2];
    const float* w2  = (const float*)d_in[3];
    float* out = (float*)d_out;

    // workspace layout
    int*   cnt  = (int*)d_ws;
    int*   tids = (int*)((char*)d_ws + 256);
    float* twts = (float*)((char*)d_ws + 256 + (size_t)NE * TOK * sizeof(int));
    size_t base = 256 + (size_t)NE * TOK * 8;            // counts + lists (~2 MB)

    const size_t n_w13 = (size_t)NE * 2 * IDIM * HID;    // 92.3M
    const size_t n_w2  = (size_t)NE * HID * IDIM;        // 46.1M
    const size_t n_hs  = (size_t)TOK * HID;              // 33.6M
    const size_t need  = base + (n_w13 + n_w2 + n_hs) * 2;

    hipMemsetAsync(cnt, 0, 256, stream);
    hipMemsetAsync(out, 0, (size_t)TOK * HID * sizeof(float), stream);

    moe_router<<<TOK / 256, 256, 0, stream>>>(lg, cnt, tids, twts);

    dim3 grid(TOK / BM, NE);
    if (ws_size >= need) {
        unsigned short* w13b = (unsigned short*)((char*)d_ws + base);
        unsigned short* w2b  = w13b + n_w13;
        unsigned short* hsb  = w2b + n_w2;
        cvt_bf16<<<8192, 256, 0, stream>>>(w13, (unsigned int*)w13b, (long long)(n_w13 / 4));
        cvt_bf16<<<8192, 256, 0, stream>>>(w2,  (unsigned int*)w2b,  (long long)(n_w2  / 4));
        cvt_bf16<<<4096, 256, 0, stream>>>(hs,  (unsigned int*)hsb,  (long long)(n_hs  / 4));
        moe_expert<true><<<grid, 256, 0, stream>>>(hs, hsb, w13, w13b, w2, w2b,
                                                   cnt, tids, twts, out);
    } else {
        moe_expert<false><<<grid, 256, 0, stream>>>(hs, nullptr, w13, nullptr, w2, nullptr,
                                                    cnt, tids, twts, out);
    }
}
